// MambaBlock_43241730737078
// MI455X (gfx1250) — compile-verified, matrix operand order checked by reading, not run on hardware
//
#include <hip/hip_runtime.h>
#include <hip/hip_bf16.h>

// ---------------------------------------------------------------------------
// Mamba block for MI455X (gfx1250): bf16 WMMA GEMMs (TDM-staged LDS tiles) +
// fp32 VALU selective scan.
// ---------------------------------------------------------------------------

#define DIM    1024
#define STATE  16
#define KCONV  4
#define INNER  2048
#define BATCH  4
#define SEQ    2048
#define EPSV   1e-6f

typedef __attribute__((ext_vector_type(16))) __bf16 bf16x16;
typedef __attribute__((ext_vector_type(8)))  __bf16 bf16x8;
typedef __attribute__((ext_vector_type(8)))  float  f32x8;

typedef __attribute__((ext_vector_type(4))) unsigned tdm_u32x4;
typedef __attribute__((ext_vector_type(8))) int      tdm_i32x8;
typedef __attribute__((ext_vector_type(4))) int      tdm_i32x4;

#if defined(__has_builtin)
#if __has_builtin(__builtin_amdgcn_tensor_load_to_lds)
#define HAVE_TDM 1
#endif
#endif

// float -> bf16 (round-to-nearest-even), stored as ushort
static __device__ __forceinline__ unsigned short f2bf(float f) {
  unsigned int u = __float_as_uint(f);
  u += 0x7FFFu + ((u >> 16) & 1u);
  return (unsigned short)(u >> 16);
}
static __device__ __forceinline__ float bf2f(unsigned short v) {
  return __uint_as_float(((unsigned int)v) << 16);
}
static __device__ __forceinline__ float siluf(float v) {
  return v / (1.0f + __expf(-v));
}
static __device__ __forceinline__ float softplusf(float v) {
  return v > 20.0f ? v : log1pf(__expf(v));
}

#if defined(HAVE_TDM)
// Issue one 2-D TDM tile load: tile_d1 rows x tile_d0 bf16 elems from a
// row-major matrix (row stride = stride_elems) into contiguous LDS at lds_off.
// D# packing per CDNA5 ISA.
// NOTE: this toolchain (clang-23 / therock-10.0) declares the 6-arg form:
//   (u32x4 g0, i32x8 g1, i32x4 g2, i32x4 g3, i32x8 g4, i32 cpol)
static __device__ __forceinline__ void tdm_load_tile_2d(
    unsigned lds_off, const unsigned short* gptr, unsigned tensor_d0,
    unsigned tensor_d1, unsigned tile_d0, unsigned tile_d1,
    unsigned stride_elems) {
  const unsigned long long ga = (unsigned long long)(uintptr_t)gptr;
  tdm_u32x4 g0;
  g0[0] = 1u;                                                // count=1, user D#
  g0[1] = lds_off;                                           // lds_addr
  g0[2] = (unsigned)(ga & 0xFFFFFFFFull);                    // global_addr lo
  g0[3] = (unsigned)((ga >> 32) & 0x01FFFFFFull) | (2u << 30);  // hi | type=2
  tdm_i32x8 g1;
  g1[0] = (int)(1u << 16);                    // data_size=2B; no multicast
  g1[1] = (int)((tensor_d0 & 0xFFFFu) << 16); // tensor_dim0[15:0]
  g1[2] = (int)((tensor_d0 >> 16) | ((tensor_d1 & 0xFFFFu) << 16));
  g1[3] = (int)((tensor_d1 >> 16) | (tile_d0 << 16));        // tile_dim0
  g1[4] = (int)(tile_d1 & 0xFFFFu);                          // tile_dim1; dim2=0
  g1[5] = (int)stride_elems;                                 // dim0_stride lo
  g1[6] = 0;                                                 // stride hi = 0
  g1[7] = 0;
  const tdm_i32x4 z4 = {0, 0, 0, 0};
  const tdm_i32x8 z8 = {0, 0, 0, 0, 0, 0, 0, 0};
  __builtin_amdgcn_tensor_load_to_lds(g0, g1, z4, z4, z8, 0);
}
#endif

// ---------------------------------------------------------------------------
// RMSNorm over DIM, write bf16. One block (256 thr) per row of 1024.
// ---------------------------------------------------------------------------
__global__ __launch_bounds__(256) void rmsnorm_bf16_kernel(
    const float* __restrict__ x, const float* __restrict__ w,
    unsigned short* __restrict__ out) {
  const int r = blockIdx.x;
  const int tid = threadIdx.x;
  float v[4];
  float ss = 0.f;
#pragma unroll
  for (int k = 0; k < 4; ++k) {
    v[k] = x[(size_t)r * DIM + tid + k * 256];
    ss += v[k] * v[k];
  }
  __shared__ float red[256];
  red[tid] = ss;
  __syncthreads();
  for (int s = 128; s > 0; s >>= 1) {
    if (tid < s) red[tid] += red[tid + s];
    __syncthreads();
  }
  const float scale = rsqrtf(red[0] * (1.0f / DIM) + EPSV);
#pragma unroll
  for (int k = 0; k < 4; ++k) {
    const int c = tid + k * 256;
    out[(size_t)r * DIM + c] = f2bf(v[k] * scale * w[c]);
  }
}

// ---------------------------------------------------------------------------
// Elementwise f32 -> bf16 cast (weights).
// ---------------------------------------------------------------------------
__global__ __launch_bounds__(256) void cast_bf16_kernel(
    const float* __restrict__ in, unsigned short* __restrict__ out, int n) {
  const int i = blockIdx.x * 256 + threadIdx.x;
  if (i < n) out[i] = f2bf(in[i]);
}

// ---------------------------------------------------------------------------
// Tiled bf16 WMMA GEMM:  Out[M,N] = X[M,K] * W[N,K]^T  (both row-major bf16)
// Block tile 128(M) x 64(N), K-step 32. 8 waves: 4 along M x 2 along N,
// each wave computes a 32x32 tile = four v_wmma_f32_16x16x32_bf16 accums.
// FULL=true: exact tiles, TDM tensor_load_to_lds staging (wave 0 issues the
// A/B tile DMAs, waits on TENSORcnt, workgroup barrier publishes LDS).
// FULL=false: guarded per-thread staging (used for the N=32 B/C projection).
// Epilogue: optional bias, optional softplus, column split into out0/out1.
// ---------------------------------------------------------------------------
#define TM 128
#define TN 64
#define TK 32

template <bool FULL>
__global__ __launch_bounds__(256) void gemm_bf16_wmma_kernel(
    const unsigned short* __restrict__ X,  // M x K bf16
    const unsigned short* __restrict__ W,  // N x K bf16
    float* __restrict__ out0, float* __restrict__ out1,
    int M, int N, int K, int splitN,
    const float* __restrict__ bias, int act) {
  __shared__ __align__(16) unsigned short lA[TM * TK];  // 8 KB
  __shared__ __align__(16) unsigned short lB[TN * TK];  // 4 KB

  const int tid  = threadIdx.x;
  const int lane = tid & 31;
  const int wave = tid >> 5;
  const int half = lane >> 4;   // 0: K 0-7 & 16-23, 1: K 8-15 & 24-31
  const int l16  = lane & 15;

  const int rowBase = blockIdx.y * TM;
  const int colBase = blockIdx.x * TN;
  const int wM = (wave & 3) * 32;
  const int wN = (wave >> 2) * 32;

  const f32x8 zero = {0.f, 0.f, 0.f, 0.f, 0.f, 0.f, 0.f, 0.f};
  f32x8 acc[2][2];
#pragma unroll
  for (int mi = 0; mi < 2; ++mi)
#pragma unroll
    for (int ni = 0; ni < 2; ++ni) acc[mi][ni] = zero;

  const int kTiles = K / TK;
  for (int kt = 0; kt < kTiles; ++kt) {
    // ---------------- stage A/B tiles into LDS ----------------
    if (FULL) {
#if defined(HAVE_TDM)
      if (wave == 0) {
        tdm_load_tile_2d((unsigned)(uintptr_t)&lA[0],
                         X + (size_t)rowBase * K + (size_t)kt * TK,
                         (unsigned)K, (unsigned)M, TK, TM, (unsigned)K);
        tdm_load_tile_2d((unsigned)(uintptr_t)&lB[0],
                         W + (size_t)colBase * K + (size_t)kt * TK,
                         (unsigned)K, (unsigned)N, TK, TN, (unsigned)K);
        __builtin_amdgcn_s_wait_tensorcnt(0);
      }
#else
      // unguarded per-thread staging (exact tiles)
      int chunk = tid;
#pragma unroll
      for (int it = 0; it < 2; ++it, chunk += 256) {
        const int r = chunk >> 2, c8 = (chunk & 3) * 8;
        *(uint4*)(lA + r * TK + c8) =
            *(const uint4*)(X + (size_t)(rowBase + r) * K + kt * TK + c8);
      }
      const int r = tid >> 2, c8 = (tid & 3) * 8;
      *(uint4*)(lB + r * TK + c8) =
          *(const uint4*)(W + (size_t)(colBase + r) * K + kt * TK + c8);
#endif
    } else {
      // guarded per-thread staging (N may not be a multiple of TN)
      int chunk = tid;
#pragma unroll
      for (int it = 0; it < 2; ++it, chunk += 256) {
        const int r = chunk >> 2, c8 = (chunk & 3) * 8;
        const int gr = rowBase + r;
        uint4 v = make_uint4(0u, 0u, 0u, 0u);
        if (gr < M) v = *(const uint4*)(X + (size_t)gr * K + kt * TK + c8);
        *(uint4*)(lA + r * TK + c8) = v;
      }
      const int r = tid >> 2, c8 = (tid & 3) * 8;
      const int gn = colBase + r;
      uint4 v = make_uint4(0u, 0u, 0u, 0u);
      if (gn < N) v = *(const uint4*)(W + (size_t)gn * K + kt * TK + c8);
      *(uint4*)(lB + r * TK + c8) = v;
    }
    __syncthreads();

    // ---------------- fragments per 16-bit A/B lane layout ----------------
    bf16x16 afrag[2], bfrag[2];
#pragma unroll
    for (int mi = 0; mi < 2; ++mi) {
      const unsigned short* p = lA + (wM + mi * 16 + l16) * TK + half * 8;
      bf16x8 lo = *(const bf16x8*)p;
      bf16x8 hi = *(const bf16x8*)(p + 16);
      afrag[mi] = __builtin_shufflevector(lo, hi, 0, 1, 2, 3, 4, 5, 6, 7, 8, 9,
                                          10, 11, 12, 13, 14, 15);
    }
#pragma unroll
    for (int ni = 0; ni < 2; ++ni) {
      const unsigned short* p = lB + (wN + ni * 16 + l16) * TK + half * 8;
      bf16x8 lo = *(const bf16x8*)p;
      bf16x8 hi = *(const bf16x8*)(p + 16);
      bfrag[ni] = __builtin_shufflevector(lo, hi, 0, 1, 2, 3, 4, 5, 6, 7, 8, 9,
                                          10, 11, 12, 13, 14, 15);
    }

#pragma unroll
    for (int mi = 0; mi < 2; ++mi)
#pragma unroll
      for (int ni = 0; ni < 2; ++ni)
        acc[mi][ni] = __builtin_amdgcn_wmma_f32_16x16x32_bf16(
            false, afrag[mi], false, bfrag[ni], (short)0, acc[mi][ni], false,
            false);
    __syncthreads();
  }

  // ---------------- epilogue: D layout => M = v + 8*half, N = l16 ----------
  const int ld1 = N - splitN;
#pragma unroll
  for (int mi = 0; mi < 2; ++mi)
#pragma unroll
    for (int ni = 0; ni < 2; ++ni)
#pragma unroll
      for (int v = 0; v < 8; ++v) {
        const int row = rowBase + wM + mi * 16 + v + 8 * half;
        const int col = colBase + wN + ni * 16 + l16;
        if (FULL || (row < M && col < N)) {
          float val = acc[mi][ni][v];
          if (bias) val += bias[col];
          if (act == 1) val = softplusf(val);
          if (col < splitN)
            out0[(size_t)row * splitN + col] = val;
          else
            out1[(size_t)row * ld1 + (col - splitN)] = val;
        }
      }
}

// ---------------------------------------------------------------------------
// Causal depthwise conv1d (K=4) + SiLU -> bf16. One thread per (b,t,i).
// ---------------------------------------------------------------------------
__global__ __launch_bounds__(256) void conv_silu_kernel(
    const float* __restrict__ xin,          // (B,SEQ,INNER) f32
    const float* __restrict__ cw,           // (INNER,K)
    const float* __restrict__ cb,           // (INNER)
    unsigned short* __restrict__ xconv) {   // (B,SEQ,INNER) bf16
  const size_t idx = (size_t)blockIdx.x * 256 + threadIdx.x;
  const int i = (int)(idx % INNER);
  const int t = (int)((idx / INNER) % SEQ);
  const int b = (int)(idx / ((size_t)INNER * SEQ));
  float acc = cb[i];
#pragma unroll
  for (int j = 0; j < KCONV; ++j) {
    const int tt = t + j - (KCONV - 1);
    if (tt >= 0)
      acc += xin[((size_t)b * SEQ + tt) * INNER + i] * cw[i * KCONV + j];
  }
  xconv[idx] = f2bf(siluf(acc));
}

// ---------------------------------------------------------------------------
// Selective scan. One thread per (b, channel i): h[16] in registers, serial
// over SEQ. B/C vectors staged 8 timesteps at a time through LDS. Gating
// y * silu(res) fused; bf16 result written in-place over xconv buffer.
// ---------------------------------------------------------------------------
__global__ __launch_bounds__(256) void scan_kernel(
    const float* __restrict__ dt,            // (B,SEQ,INNER)
    unsigned short* __restrict__ xconv,      // in: x_conv bf16, out: gated y
    const float* __restrict__ Bm,            // (B,SEQ,STATE)
    const float* __restrict__ Cm,            // (B,SEQ,STATE)
    const float* __restrict__ A_log,         // (INNER,STATE)
    const float* __restrict__ Dv,            // (INNER)
    const float* __restrict__ res) {         // (B,SEQ,INNER)
  const int tid = threadIdx.x;
  const int gid = blockIdx.x * 256 + tid;
  const int b = gid >> 11;          // / INNER
  const int i = gid & (INNER - 1);  // % INNER

  float a[STATE], h[STATE];
#pragma unroll
  for (int s = 0; s < STATE; ++s) {
    a[s] = -__expf(A_log[(size_t)i * STATE + s]);
    h[s] = 0.f;
  }
  const float Di = Dv[i];

  __shared__ float sB[8][STATE];
  __shared__ float sC[8][STATE];
  const int step = tid >> 5;  // 0..7
  const int sl = tid & 31;

  for (int t0 = 0; t0 < SEQ; t0 += 8) {
    __syncthreads();
    const size_t bc = ((size_t)b * SEQ + t0 + step) * STATE;
    if (sl < STATE)
      sB[step][sl] = Bm[bc + sl];
    else
      sC[step][sl - STATE] = Cm[bc + (sl - STATE)];
    __syncthreads();
#pragma unroll 1
    for (int u = 0; u < 8; ++u) {
      const size_t off = ((size_t)b * SEQ + t0 + u) * INNER + i;
      const float dtv = dt[off];
      const float xcv = bf2f(xconv[off]);
      float yv = Di * xcv;
#pragma unroll
      for (int s = 0; s < STATE; ++s) {
        const float dA = __expf(a[s] * dtv);
        h[s] = dA * h[s] + (dtv * sB[u][s]) * xcv;
        yv += h[s] * sC[u][s];
      }
      xconv[off] = f2bf(yv * siluf(res[off]));  // fused gating, in-place
    }
  }
}

// ---------------------------------------------------------------------------
// Host-side orchestration.
// ---------------------------------------------------------------------------
extern "C" void kernel_launch(void* const* d_in, const int* in_sizes, int n_in,
                              void* d_out, int out_size, void* d_ws,
                              size_t ws_size, hipStream_t stream) {
  (void)in_sizes; (void)n_in; (void)out_size; (void)ws_size;
  const float* x      = (const float*)d_in[0];
  const float* w_norm = (const float*)d_in[1];
  const float* W_in   = (const float*)d_in[2];
  const float* conv_w = (const float*)d_in[3];
  const float* conv_b = (const float*)d_in[4];
  const float* W_dt   = (const float*)d_in[5];
  const float* b_dt   = (const float*)d_in[6];
  const float* W_B    = (const float*)d_in[7];
  const float* W_C    = (const float*)d_in[8];
  const float* A_log  = (const float*)d_in[9];
  const float* Dv     = (const float*)d_in[10];
  const float* W_out  = (const float*)d_in[11];
  float* out = (float*)d_out;

  const int M = BATCH * SEQ;  // 8192

  // Workspace layout (bytes)
  char* base = (char*)d_ws;
  unsigned short* xn   = (unsigned short*)(base);                       // 16 MB
  unsigned short* Wb   = (unsigned short*)(base + 16777216);            //  8 MB
  unsigned short* Wbc  = (unsigned short*)(base + 16777216 + 8388608);  // 128 KB
  float* xi            = (float*)(base + 25296896);                     // 64 MB (later: dt)
  float* res           = (float*)(base + 25296896 + 67108864);          // 64 MB
  unsigned short* xc   = (unsigned short*)(base + 25296896 + 134217728);// 32 MB (later: gated y)
  float* bc            = (float*)(base + 25296896 + 167772160);         //  1 MB (Bm|Cm)
  float* dt  = xi;                  // reuse: x_inner dead after conv
  float* Bm  = bc;
  float* Cm  = bc + (size_t)M * STATE;

  // 1) RMSNorm -> bf16
  rmsnorm_bf16_kernel<<<M, 256, 0, stream>>>(x, w_norm, xn);

  // 2) cast W_in; GEMM1: (8192x1024)x(4096x1024)^T -> split x_inner | res
  cast_bf16_kernel<<<(2 * INNER * DIM + 255) / 256, 256, 0, stream>>>(
      W_in, Wb, 2 * INNER * DIM);
  {
    dim3 g((2 * INNER) / TN, M / TM);
    gemm_bf16_wmma_kernel<true><<<g, 256, 0, stream>>>(
        xn, Wb, xi, res, M, 2 * INNER, DIM, INNER, nullptr, 0);
  }

  // 3) depthwise conv + SiLU -> bf16
  conv_silu_kernel<<<((size_t)M * INNER) / 256, 256, 0, stream>>>(xi, conv_w,
                                                                  conv_b, xc);

  // 4) cast W_dt and W_B|W_C
  cast_bf16_kernel<<<(INNER * INNER + 255) / 256, 256, 0, stream>>>(
      W_dt, Wb, INNER * INNER);
  cast_bf16_kernel<<<(STATE * INNER + 255) / 256, 256, 0, stream>>>(
      W_B, Wbc, STATE * INNER);
  cast_bf16_kernel<<<(STATE * INNER + 255) / 256, 256, 0, stream>>>(
      W_C, Wbc + (size_t)STATE * INNER, STATE * INNER);

  // 5) GEMM2: dt = softplus(x_conv @ W_dt^T + b_dt)   (overwrites x_inner)
  {
    dim3 g(INNER / TN, M / TM);
    gemm_bf16_wmma_kernel<true><<<g, 256, 0, stream>>>(
        xc, Wb, dt, dt, M, INNER, INNER, INNER, b_dt, 1);
  }
  // 6) GEMM_BC: (8192x2048)x(32x2048)^T -> split Bm | Cm (bounded path, N=32)
  {
    dim3 g((2 * STATE + TN - 1) / TN, M / TM);
    gemm_bf16_wmma_kernel<false><<<g, 256, 0, stream>>>(
        xc, Wbc, Bm, Cm, M, 2 * STATE, INNER, STATE, nullptr, 0);
  }

  // 7) selective scan + fused gating (writes bf16 gated y over xc)
  scan_kernel<<<(BATCH * INNER) / 256, 256, 0, stream>>>(dt, xc, Bm, Cm, A_log,
                                                         Dv, res);

  // 8) cast W_out; GEMM3: (8192x2048)x(1024x2048)^T -> d_out (f32)
  cast_bf16_kernel<<<(DIM * INNER + 255) / 256, 256, 0, stream>>>(
      W_out, Wb, DIM * INNER);
  {
    dim3 g(DIM / TN, M / TM);
    gemm_bf16_wmma_kernel<true><<<g, 256, 0, stream>>>(
        xc, Wb, out, out, M, DIM, INNER, DIM, nullptr, 0);
  }
}
